// Meta_Module_61641370632880
// MI455X (gfx1250) — matrix operand order repaired and checked by reading, not compiled, not run on hardware
//
#include <hip/hip_runtime.h>
#include <math.h>
#include <cstdint>

// CDNA5 / gfx1250 WMMA types
typedef __attribute__((ext_vector_type(16))) __bf16 v16bf;
typedef __attribute__((ext_vector_type(8)))  float  v8f;
typedef __attribute__((ext_vector_type(4)))  int    v4i;

// Pointers for the async global->LDS builtin:
//   arg0: int4 in addrspace(1) (global), arg1: int4 in addrspace(3) (LDS)
typedef __attribute__((address_space(1))) v4i* as1_v4i_ptr;
typedef __attribute__((address_space(3))) v4i* as3_v4i_ptr;

constexpr int kB  = 512;   // batch
constexpr int kS  = 200;   // seq
constexpr int kSP = 208;   // seq padded to 13*16
constexpr int kH  = 256;   // hidden
constexpr int kMT = 13;    // M tiles (16 rows each)

#if __has_builtin(__builtin_amdgcn_global_load_async_to_lds_b128)
#define HAVE_ASYNC_LDS 1
#else
#define HAVE_ASYNC_LDS 0
#endif

// ---------------------------------------------------------------------------
// Pre-pass: convert U (f32 [H][H], row-major, a_k = sum_h mem_h * U[k][h]) to
// bf16 in workspace. B-matrix fragments then read contiguous rows of U.
// ---------------------------------------------------------------------------
__global__ void cvt_u_bf16(const float* __restrict__ U,
                           unsigned short* __restrict__ Ub) {
    int i = blockIdx.x * blockDim.x + threadIdx.x;   // 65536 total
    unsigned u = __float_as_uint(U[i]);
    u += 0x7fffu + ((u >> 16) & 1u);                 // round-to-nearest-even
    Ub[i] = (unsigned short)(u >> 16);
}

// ---------------------------------------------------------------------------
// Fused additive-attention kernel: one workgroup per batch element.
//   scores[s] = V . tanh(mem[s] @ U^T + last @ W^T)
//   out       = softmax(scores) . mem @ MetaW^T + Metab
// ---------------------------------------------------------------------------
__launch_bounds__(256, 1)
__global__ void meta_attn_kernel(const float* __restrict__ mem_g,
                                 const float* __restrict__ last_g,
                                 const unsigned short* __restrict__ Ub,
                                 const float* __restrict__ Wm,
                                 const float* __restrict__ Vv,
                                 const float* __restrict__ MetaW,
                                 const float* __restrict__ Metab,
                                 float* __restrict__ out) {
    extern __shared__ char smem_raw[];
    float* smem   = (float*)smem_raw;     // [kSP*kH] staged activations (f32)
    float* lastm  = smem + kSP * kH;      // [kH]
    float* lvec   = lastm + kH;           // [kH]  l = last @ W^T
    float* vvec   = lvec + kH;            // [kH]  V
    float* scores = vvec + kH;            // [256] (padded)
    float* pooled = scores + 256;         // [kH]
    float* red    = pooled + kH;          // [16]  reduction scratch

    const int tid  = threadIdx.x;
    const int b    = blockIdx.x;
    const int wid  = tid >> 5;
    const int lane = tid & 31;
    const int lh   = lane >> 4;           // lane half (0/1)
    const int l16  = lane & 15;

    // --- stage all_memory[b] (200x256 f32) into LDS -------------------------
#if HAVE_ASYNC_LDS
    {
        // Fire-and-forget DMA-style copies (ASYNCcnt); no VGPR transit.
        // Dynamic LDS starts at byte offset 0 (no static __shared__ here).
        const char* gbase = (const char*)(mem_g + (size_t)b * kS * kH);
        #pragma unroll
        for (int i = 0; i < (kS * kH * 4) / (256 * 16); ++i) {   // 50 iters
            unsigned off = (unsigned)(tid + i * 256) * 16u;
            __builtin_amdgcn_global_load_async_to_lds_b128(
                (as1_v4i_ptr)(uintptr_t)(gbase + off),
                (as3_v4i_ptr)(uintptr_t)off,
                0, 0);
        }
    }
#else
    {
        const float4* g4 = (const float4*)(mem_g + (size_t)b * kS * kH);
        float4* s4 = (float4*)smem;
        #pragma unroll
        for (int i = 0; i < (kS * kH / 4) / 256; ++i)            // 50 float4
            s4[tid + i * 256] = g4[tid + i * 256];
    }
#endif
    // zero-pad rows 200..207 (normal DS stores, independent LDS region)
    {
        float4 z = {0.f, 0.f, 0.f, 0.f};
        float4* p4 = (float4*)(smem + kS * kH);                  // 512 float4
        p4[tid * 2]     = z;
        p4[tid * 2 + 1] = z;
    }

    // warm L2 with the bf16 U matrix while the copies are in flight
    {
        const char* up = (const char*)Ub + (size_t)tid * 512;    // covers 128 KB
        __builtin_prefetch(up,       0, 0);
        __builtin_prefetch(up + 128, 0, 0);
        __builtin_prefetch(up + 256, 0, 0);
        __builtin_prefetch(up + 384, 0, 0);
    }

    // --- stage small vectors ---
    lastm[tid] = last_g[(size_t)b * kH + tid];
    vvec[tid]  = Vv[tid];
    __syncthreads();   // lastm visible (async copies still in flight)

    // --- l[k] = sum_h lastm[h] * W[k][h] — overlaps with async staging ---
    {
        const float4* wrow = (const float4*)(Wm + (size_t)tid * kH);
        const float4* l4   = (const float4*)lastm;
        float acc = 0.f;
        #pragma unroll 8
        for (int i = 0; i < kH / 4; ++i) {
            float4 w = wrow[i], x = l4[i];
            acc += w.x * x.x + w.y * x.y + w.z * x.z + w.w * x.w;
        }
        lvec[tid] = acc;
    }

#if HAVE_ASYNC_LDS
#if __has_builtin(__builtin_amdgcn_s_wait_asynccnt)
    __builtin_amdgcn_s_wait_asynccnt(0);
#else
    asm volatile("s_wait_asynccnt 0" ::: "memory");
#endif
#endif
    __syncthreads();   // staged activations + lvec visible to all waves

    // --- main GEMM: a = mem @ U^T via bf16 WMMA, fused tanh/V epilogue ---
    for (int m = wid; m < kMT; m += 8) {
        const int srow = m * 16 + l16;      // A row supplied by this lane
        float sacc[8];
        #pragma unroll
        for (int r = 0; r < 8; ++r) sacc[r] = 0.f;

        for (int nt = 0; nt < 16; ++nt) {
            const int n = nt * 16 + l16;    // output column (both halves)
            v8f c = {};
            #pragma unroll
            for (int kt = 0; kt < 8; ++kt) {
                const int k0 = kt * 32;
                // A fragment: 16-bit A 16x32 layout — lane half selects
                // K = k0 + lh*8 + {0..7} and k0 + 16 + lh*8 + {0..7}
                const float* rp = smem + srow * kH + k0 + lh * 8;
                float4 f0 = *(const float4*)(rp);
                float4 f1 = *(const float4*)(rp + 4);
                float4 f2 = *(const float4*)(rp + 16);
                float4 f3 = *(const float4*)(rp + 20);
                v16bf av;
                av[0]=(__bf16)f0.x;  av[1]=(__bf16)f0.y;  av[2]=(__bf16)f0.z;  av[3]=(__bf16)f0.w;
                av[4]=(__bf16)f1.x;  av[5]=(__bf16)f1.y;  av[6]=(__bf16)f1.z;  av[7]=(__bf16)f1.w;
                av[8]=(__bf16)f2.x;  av[9]=(__bf16)f2.y;  av[10]=(__bf16)f2.z; av[11]=(__bf16)f2.w;
                av[12]=(__bf16)f3.x; av[13]=(__bf16)f3.y; av[14]=(__bf16)f3.z; av[15]=(__bf16)f3.w;
                // B fragment: B[k][n] = U[n][k]; lane n reads bf16 of row n
                // at K = k0 + lh*8 + {0..7} and k0 + 16 + lh*8 + {0..7}
                union { uint4 q[2]; v16bf v; } bu;
                const unsigned short* bp = Ub + (size_t)n * kH + k0 + lh * 8;
                bu.q[0] = *(const uint4*)(bp);
                bu.q[1] = *(const uint4*)(bp + 16);
                c = __builtin_amdgcn_wmma_f32_16x16x32_bf16(
                        false, av, false, bu.v, (short)0, c, false, false);
            }
            // epilogue: accumulate tanh(a + l[n]) * V[n] per row
            const float lv = lvec[n];
            const float vw = vvec[n];
            #pragma unroll
            for (int r = 0; r < 8; ++r) {
                float x = c[r] + lv;
#if __has_builtin(__builtin_amdgcn_tanhf)
                x = __builtin_amdgcn_tanhf(x);
#else
                x = tanhf(x);
#endif
                sacc[r] += x * vw;
            }
        }
        // reduce over N within each 16-lane half (D row = m*16 + r + lh*8)
        #pragma unroll
        for (int r = 0; r < 8; ++r) {
            float v = sacc[r];
            v += __shfl_xor(v, 1, 32);
            v += __shfl_xor(v, 2, 32);
            v += __shfl_xor(v, 4, 32);
            v += __shfl_xor(v, 8, 32);
            if (l16 == 0) scores[m * 16 + r + lh * 8] = v;
        }
    }
    __syncthreads();
    if (tid >= kS) scores[tid] = -INFINITY;      // mask pad rows
    __syncthreads();

    // --- softmax over scores[0..199] ---
    const float sc = scores[tid];
    float mx = sc;
    #pragma unroll
    for (int off = 16; off; off >>= 1) mx = fmaxf(mx, __shfl_xor(mx, off, 32));
    if (lane == 0) red[wid] = mx;
    __syncthreads();
    if (tid == 0) {
        float g = red[0];
        for (int i = 1; i < 8; ++i) g = fmaxf(g, red[i]);
        red[8] = g;
    }
    __syncthreads();
    const float gmax = red[8];
    const float e = (tid < kS) ? __expf(sc - gmax) : 0.f;
    scores[tid] = e;
    float ss = e;
    #pragma unroll
    for (int off = 16; off; off >>= 1) ss += __shfl_xor(ss, off, 32);
    if (lane == 0) red[wid] = ss;
    __syncthreads();
    if (tid == 0) {
        float t = 0.f;
        for (int i = 0; i < 8; ++i) t += red[i];
        red[9] = t;
    }
    __syncthreads();
    const float inv = 1.f / red[9];

    // --- pooled[h] = sum_s alpha_s * mem[s][h] (reuse LDS-staged activations)
    float acc = 0.f;
    for (int s = 0; s < kS; ++s)
        acc = fmaf(scores[s], smem[s * kH + tid], acc);
    pooled[tid] = acc * inv;
    __syncthreads();

    // --- out[b][j] = pooled . MetaW[j] + Metab[j] ---
    if (tid < 4) {
        const float* mrow = MetaW + tid * kH;
        float o = Metab[tid];
        for (int h2 = 0; h2 < kH; ++h2) o = fmaf(pooled[h2], mrow[h2], o);
        out[b * 4 + tid] = o;
    }
}

// ---------------------------------------------------------------------------
extern "C" void kernel_launch(void* const* d_in, const int* in_sizes, int n_in,
                              void* d_out, int out_size, void* d_ws, size_t ws_size,
                              hipStream_t stream) {
    const float* all_memory  = (const float*)d_in[0];
    const float* last_memory = (const float*)d_in[1];
    const float* U           = (const float*)d_in[2];
    const float* W           = (const float*)d_in[3];
    const float* V           = (const float*)d_in[4];
    const float* MetaW       = (const float*)d_in[5];
    const float* Metab       = (const float*)d_in[6];
    float* out               = (float*)d_out;

    unsigned short* Ub = (unsigned short*)d_ws;     // 65536 * 2 B = 128 KB

    cvt_u_bf16<<<(kH * kH) / 256, 256, 0, stream>>>(U, Ub);

    // LDS: 208*256*4 (acts) + 4*256*4 (lastm,lvec,vvec,pooled) + 256*4 + 64
    const size_t lds_bytes = (size_t)kSP * kH * 4 + 4 * kH * 4 + 256 * 4 + 64;
    meta_attn_kernel<<<kB, 256, lds_bytes, stream>>>(
        all_memory, last_memory, Ub, W, V, MetaW, Metab, out);
}
